// NonLocalBlock_NLRN_5257039970607
// MI455X (gfx1250) — matrix operand order repaired and checked by reading, not compiled, not run on hardware
//
#include <hip/hip_runtime.h>
#include <hip/hip_bf16.h>

// ---------------- CDNA5 WMMA types ----------------
typedef __attribute__((ext_vector_type(16))) __bf16 v16bf;
typedef __attribute__((ext_vector_type(8)))  float  v8f;
typedef __attribute__((address_space(3)))    float  lds_float;

static __device__ __forceinline__ void cvt4(v16bf& d, int i, float4 f) {
    d[i + 0] = (__bf16)f.x; d[i + 1] = (__bf16)f.y;
    d[i + 2] = (__bf16)f.z; d[i + 3] = (__bf16)f.w;
}

// A-fragment element->K mapping for 16-bit A 16x32 (per lane half)
static __device__ __forceinline__ int koff(int e) { return (e < 8) ? e : e + 8; }

// byte offset of a __shared__ object within the wave's LDS aperture
static __device__ __forceinline__ unsigned lds_off(const void* p) {
    return (unsigned)(size_t)(lds_float*)p;
}

// CDNA5 async copy: 16B per lane, global -> LDS, tracked by ASYNCcnt
static __device__ __forceinline__ void async_load_b128(unsigned lds_byte_off,
                                                       const float* gaddr) {
    asm volatile("global_load_async_to_lds_b128 %0, %1, off"
                 :: "v"(lds_byte_off), "v"((unsigned long long)(size_t)gaddr)
                 : "memory");
}
static __device__ __forceinline__ void async_wait0() {
    asm volatile("s_wait_asynccnt 0x0" ::: "memory");
}

// Problem constants (fixed by the reference)
// x: [4,256,64,64]  w_phi/w_theta/w_g: [128,256]  w_mask: [256,128]
#define BATCH 4
#define CIN   256
#define CI    128
#define HW    64
#define PIX   4096          // 64*64 per batch
#define NPIX  16384         // 4*4096
#define KK    49
#define FLATB 6272          // CI*KK, per-(h,w) flat stride of the scrambled reshape

#define APADQ 20            // LDS stride (floats) for [k][m] X tile (bank-safe)
#define APADM 132           // LDS stride (floats) for [m][ci] yout tile (16B-aligned)

// =====================================================================
// Kernel 1: fused phi/theta/g 1x1 convs as bf16 WMMA GEMM.
// Block = one 16-pixel tile; 8 waves cover the 8 ci-out tiles.
// X tile (16 px x 256 ch) async-staged to LDS once per block.
// =====================================================================
__global__ __launch_bounds__(256) void k_qkv(
    const float* __restrict__ x,
    const float* __restrict__ w_phi, const float* __restrict__ w_theta,
    const float* __restrict__ w_g,
    float* __restrict__ phi, float* __restrict__ theta, float* __restrict__ g)
{
    __shared__ float ldsA[CIN * APADQ];        // [k][m], padded; 20480 B

    const int lane = threadIdx.x & 31;
    const int wv   = threadIdx.x >> 5;
    const int mt   = blockIdx.x;               // pixel tile 0..1023
    const int nt   = wv;                       // ci tile 0..7
    const int p0   = mt << 4;
    const int bb   = p0 >> 12;
    const int rowb = p0 & 4095;                // y*64 + x0 (tiles never cross rows)
    const int m    = lane & 15;
    const int hi   = lane >> 4;                // lane-half
    const int n    = m;
    const int n0   = nt << 4;

    // ---- async stage: wave wv loads channel rows [wv*32, wv*32+32) ----
    {
        const unsigned lbase = lds_off(ldsA);
        const int r0 = wv * 32 + (lane >> 2);      // +8 per iteration
        const int c4 = (lane & 3) * 4;             // 4-float column chunk
        #pragma unroll
        for (int it = 0; it < 4; ++it) {
            const int r = r0 + it * 8;
            async_load_b128(lbase + (unsigned)(r * APADQ + c4) * 4u,
                            x + ((size_t)bb * CIN + r) * PIX + rowb + c4);
        }
        async_wait0();
        __syncthreads();
    }

    v8f accP = {}; v8f accT = {}; v8f accG = {};
    for (int k0 = 0; k0 < CIN; k0 += 32) {
        // ---- A fragment from LDS (16-bit A 16x32 layout) ----
        v16bf a;
        const int kb = k0 + hi * 8;
        #pragma unroll
        for (int e = 0; e < 16; ++e)
            a[e] = (__bf16)ldsA[(kb + koff(e)) * APADQ + m];

        // ---- B fragments from global (L2-resident weights), K = hi*8 + koff(e) ----
        const int kw = k0 + hi * 8;
        if (k0 + 32 < CIN)
            __builtin_prefetch(w_phi + (size_t)(n0 + n) * CIN + kw + 32, 0, 3);
        const float4* wp4 = (const float4*)(w_phi   + (size_t)(n0 + n) * CIN + kw);
        const float4* wt4 = (const float4*)(w_theta + (size_t)(n0 + n) * CIN + kw);
        const float4* wg4 = (const float4*)(w_g     + (size_t)(n0 + n) * CIN + kw);
        v16bf bp, bt, bg;
        #pragma unroll
        for (int q = 0; q < 4; ++q) {
            cvt4(bp, 4 * q, wp4[q + (q & 2)]);
            cvt4(bt, 4 * q, wt4[q + (q & 2)]);
            cvt4(bg, 4 * q, wg4[q + (q & 2)]);
        }

        accP = __builtin_amdgcn_wmma_f32_16x16x32_bf16(false, a, false, bp, (short)0, accP, false, false);
        accT = __builtin_amdgcn_wmma_f32_16x16x32_bf16(false, a, false, bt, (short)0, accT, false, false);
        accG = __builtin_amdgcn_wmma_f32_16x16x32_bf16(false, a, false, bg, (short)0, accG, false, false);
    }

    // ---- store D: M = v + hi*8, N = n; pixel-major [p, ci] ----
    #pragma unroll
    for (int v = 0; v < 8; ++v) {
        const size_t o = (size_t)(p0 + v + hi * 8) * CI + n0 + n;
        phi[o]   = accP[v];
        theta[o] = accT[v];
        g[o]     = accG[v];
    }
}

// Scrambled-reshape gather: per-batch flat index -> source pixel/channel in
// pixel-major phi/g arrays.  flat = (h*64+w)*6272 + j.
static __device__ __forceinline__ float gather_patch(
    const float* __restrict__ src_b, unsigned flat)
{
    const unsigned f   = flat >> 12;       // < 6272
    const unsigned r12 = flat & 4095u;
    const unsigned yy  = r12 >> 6, xx = r12 & 63u;
    const unsigned cg  = f / 49u;
    const unsigned rr  = f - cg * 49u;
    const int dy = (int)(rr / 7u) - 3;
    const int dx = (int)(rr % 7u) - 3;
    const int ny = (int)yy + dy, nx = (int)xx + dx;
    if ((unsigned)ny < 64u && (unsigned)nx < 64u)
        return src_b[(size_t)(ny * HW + nx) * CI + cg];
    return 0.0f;
}

// =====================================================================
// Kernel 2: scores[p,k] = sum_ci theta[p,ci] * PHI(base + ci*49 + k)
// one wave per pixel; lanes split ci into 4 chunks; shfl_xor reduction.
// =====================================================================
__global__ __launch_bounds__(256) void k_scores(
    const float* __restrict__ theta, const float* __restrict__ phi,
    float* __restrict__ scores)
{
    const int lane = threadIdx.x & 31;
    const int p = blockIdx.x * 8 + (threadIdx.x >> 5);   // 0..16383
    const int bb = p >> 12;
    const unsigned base = (unsigned)(p & 4095) * (unsigned)FLATB;
    const float* phib = phi + (size_t)bb * PIX * CI;

    float t[4];
    #pragma unroll
    for (int c = 0; c < 4; ++c)
        t[c] = theta[(size_t)p * CI + c * 32 + lane];

    for (int k = 0; k < KK; ++k) {
        float s = 0.0f;
        #pragma unroll
        for (int c = 0; c < 4; ++c) {
            const unsigned ci = (unsigned)(c * 32 + lane);
            s += t[c] * gather_patch(phib, base + ci * 49u + (unsigned)k);
        }
        #pragma unroll
        for (int off = 16; off >= 1; off >>= 1)
            s += __shfl_xor(s, off, 32);
        if (lane == 0) scores[(size_t)p * KK + k] = s;
    }
}

// =====================================================================
// Kernel 3: softmax over the h axis (axis=1 of [b,h,w,1,kk]) — faithful.
// block = (b, x) column, thread = k; in-place on scores.
// =====================================================================
__global__ __launch_bounds__(64) void k_softmax(float* __restrict__ sc)
{
    const int k = threadIdx.x;
    if (k >= KK) return;
    const int bb = blockIdx.x >> 6;
    const int xx = blockIdx.x & 63;
    const size_t base = ((size_t)bb * PIX + xx) * KK + k;   // +y*64*49
    const size_t ystr = (size_t)HW * KK;                    // 3136

    float mx = -3.4e38f;
    for (int y = 0; y < HW; ++y) mx = fmaxf(mx, sc[base + y * ystr]);
    float sum = 0.0f;
    for (int y = 0; y < HW; ++y) sum += __expf(sc[base + y * ystr] - mx);
    const float inv = 1.0f / sum;
    for (int y = 0; y < HW; ++y) {
        const size_t i = base + y * ystr;
        sc[i] = __expf(sc[i] - mx) * inv;
    }
}

// =====================================================================
// Kernel 4: yout[p,ci] = sum_k attn[p,k] * G(base + k*128 + ci)
// one wave per pixel; lanes over ci.
// =====================================================================
__global__ __launch_bounds__(256) void k_agg(
    const float* __restrict__ attn, const float* __restrict__ g,
    float* __restrict__ yout)
{
    const int lane = threadIdx.x & 31;
    const int p = blockIdx.x * 8 + (threadIdx.x >> 5);
    const int bb = p >> 12;
    const unsigned base = (unsigned)(p & 4095) * (unsigned)FLATB;
    const float* gb = g + (size_t)bb * PIX * CI;

    float acc[4] = {0.f, 0.f, 0.f, 0.f};
    for (int k = 0; k < KK; ++k) {
        const float a = attn[(size_t)p * KK + k];
        #pragma unroll
        for (int c = 0; c < 4; ++c) {
            const unsigned ci = (unsigned)(c * 32 + lane);
            acc[c] += a * gather_patch(gb, base + (unsigned)k * 128u + ci);
        }
    }
    #pragma unroll
    for (int c = 0; c < 4; ++c)
        yout[(size_t)p * CI + c * 32 + lane] = acc[c];
}

// =====================================================================
// Kernel 5: mask conv as WMMA GEMM [N,128]x[128,256] + residual add.
// Block = half the cout tiles of one pixel tile; yout tile async-staged
// to LDS; LDS transpose of D tile -> coalesced [b,c,h,w] stores.
// =====================================================================
__global__ __launch_bounds__(256) void k_mask(
    const float* __restrict__ yout, const float* __restrict__ w_mask,
    const float* __restrict__ x, float* __restrict__ out)
{
    __shared__ float ldsY[16 * APADM];     // [m][ci], 8448 B
    __shared__ float ldsT[8 * 16 * 17];    // per-wave transpose tiles, 8704 B

    const int lane = threadIdx.x & 31;
    const int wv   = threadIdx.x >> 5;
    const int mt   = blockIdx.x >> 1;              // pixel tile 0..1023
    const int nt   = ((blockIdx.x & 1) << 3) + wv; // cout tile 0..15
    const int p0   = mt << 4;
    const int bb   = p0 >> 12;
    const int rowb = p0 & 4095;
    const int m    = lane & 15;
    const int hi   = lane >> 4;
    const int n    = m;
    const int n0   = nt << 4;

    // ---- async stage yout tile: one instruction = one pixel's 128 ci ----
    {
        const unsigned lbase = lds_off(ldsY);
        #pragma unroll
        for (int j = 0; j < 2; ++j) {
            const int mm = wv * 2 + j;
            async_load_b128(lbase + (unsigned)(mm * APADM + 4 * lane) * 4u,
                            yout + (size_t)(p0 + mm) * CI + 4 * lane);
        }
        async_wait0();
        __syncthreads();
    }

    v8f acc = {};
    #pragma unroll
    for (int k0 = 0; k0 < CI; k0 += 32) {
        // A fragment from LDS
        v16bf a;
        const int kb = k0 + hi * 8;
        #pragma unroll
        for (int e = 0; e < 16; ++e)
            a[e] = (__bf16)ldsY[m * APADM + kb + koff(e)];

        // B: w_mask[cout, ci] contiguous in K
        const float4* w4 = (const float4*)(w_mask + (size_t)(n0 + n) * CI + k0 + hi * 8);
        v16bf bm;
        #pragma unroll
        for (int q = 0; q < 4; ++q) cvt4(bm, 4 * q, w4[q + (q & 2)]);

        acc = __builtin_amdgcn_wmma_f32_16x16x32_bf16(false, a, false, bm, (short)0, acc, false, false);
    }

    // Transpose through LDS: write [n][m], read rows of m for coalesced stores
    float* ltile = ldsT + wv * (16 * 17);
    #pragma unroll
    for (int v = 0; v < 8; ++v)
        ltile[n * 17 + v + hi * 8] = acc[v];
    __syncthreads();

    const int on = lane >> 1;          // cout within tile
    const int ms = (lane & 1) * 8;     // pixel sub-run
    const float* lrow = ltile + on * 17 + ms;
    const size_t obase = ((size_t)bb * CIN + n0 + on) * PIX + rowb + ms;
    #pragma unroll
    for (int i = 0; i < 8; ++i)
        out[obase + i] = lrow[i] + x[obase + i];
}

// =====================================================================
extern "C" void kernel_launch(void* const* d_in, const int* in_sizes, int n_in,
                              void* d_out, int out_size, void* d_ws, size_t ws_size,
                              hipStream_t stream)
{
    const float* x       = (const float*)d_in[0];
    const float* w_phi   = (const float*)d_in[1];
    const float* w_theta = (const float*)d_in[2];
    const float* w_g     = (const float*)d_in[3];
    const float* w_mask  = (const float*)d_in[4];
    float* out = (float*)d_out;

    // workspace layout (floats)
    const size_t SZ_QKV = (size_t)NPIX * CI;      // 2,097,152 each
    const size_t SZ_SC  = (size_t)NPIX * KK;      //   802,816
    const size_t need   = (3 * SZ_QKV + SZ_SC + SZ_QKV) * sizeof(float);
    if (ws_size < need) return;

    float* ws    = (float*)d_ws;
    float* theta = ws;
    float* phi   = ws + SZ_QKV;
    float* g     = ws + 2 * SZ_QKV;
    float* sc    = ws + 3 * SZ_QKV;
    float* yout  = ws + 3 * SZ_QKV + SZ_SC;

    k_qkv    <<<1024, 256, 0, stream>>>(x, w_phi, w_theta, w_g, phi, theta, g);
    k_scores <<<2048, 256, 0, stream>>>(theta, phi, sc);
    k_softmax<<< 256,  64, 0, stream>>>(sc);
    k_agg    <<<2048, 256, 0, stream>>>(sc, g, yout);
    k_mask   <<<2048, 256, 0, stream>>>(yout, w_mask, x, out);
}
